// DMAN_layer_18468359373438
// MI455X (gfx1250) — matrix operand order repaired and checked by reading, not compiled, not run on hardware
//
#include <hip/hip_runtime.h>

// Sizes from reference: B=32, S=8, T=64, D=128, M=32
#define BB 32
#define SS 8
#define TT 64
#define DD 128
#define MMEM 32
#define MT 96            // M + T
#define NSEQ 256         // B*S

typedef __attribute__((ext_vector_type(16))) _Float16 v16h;
typedef __attribute__((ext_vector_type(8)))  float    v8f;
typedef __attribute__((ext_vector_type(4)))  unsigned v4u;
typedef __attribute__((ext_vector_type(4)))  int      v4i;
typedef __attribute__((ext_vector_type(8)))  int      v8i;

struct Split16 { v16h hi; v16h lo; };

// D = A*B + C with 2-term f16 splitting: a*b ~= ah*bh + ah*bl + al*bh  (~2^-22 rel err)
__device__ __forceinline__ v8f wmma3(const Split16& a, const Split16& b, v8f c) {
  c = __builtin_amdgcn_wmma_f32_16x16x32_f16(false, a.hi, false, b.hi, (short)0, c, false, false);
  c = __builtin_amdgcn_wmma_f32_16x16x32_f16(false, a.hi, false, b.lo, (short)0, c, false, false);
  c = __builtin_amdgcn_wmma_f32_16x16x32_f16(false, a.lo, false, b.hi, (short)0, c, false, false);
  return c;
}

__device__ __forceinline__ void sp(float x, Split16& s, int e) {
  _Float16 h = (_Float16)x;
  s.hi[e] = h;
  s.lo[e] = (_Float16)(x - (float)h);
}

// A tile (16 rows x 32 K), row-major source pre-offset to (row0,k0), leading dim lda.
__device__ __forceinline__ Split16 loadA(const float* base, int lda, int lane) {
  Split16 s; int r = lane & 15, h = lane >> 4;
  const float* p = base + r * lda + h * 16;
#pragma unroll
  for (int e = 0; e < 8; ++e) sp(p[e], s, e);
#pragma unroll
  for (int e = 0; e < 8; ++e) sp(p[8 + e], s, 8 + e);
  return s;
}

// B tile (32 K x 16 cols) where logical B(k,c) = Wt[c*ld + k] (weight stored [out,in]).
__device__ __forceinline__ Split16 loadBT(const float* base, int ld, int lane) {
  Split16 s; int c = lane & 15, h = lane >> 4;
  const float* p = base + c * ld + h * 16;
#pragma unroll
  for (int e = 0; e < 16; ++e) sp(p[e], s, e);
  return s;
}

// B tile (32 K x 16 cols) where logical B(k,c) = Bm[k*ld + c] (K-major source).
__device__ __forceinline__ Split16 loadB(const float* base, int ld, int lane) {
  Split16 s; int c = lane & 15, h = lane >> 4;
  const float* p = base + (h * 16) * ld + c;
#pragma unroll
  for (int e = 0; e < 16; ++e) sp(p[e * ld], s, e);
  return s;
}

// C/D layout: elem j -> row (lane>>4)*8 + j, col lane&15.
__device__ __forceinline__ void storeC(float* C, int ldc, int lane, v8f acc, float scale) {
  int c = lane & 15, h = lane >> 4;
#pragma unroll
  for (int j = 0; j < 8; ++j) C[(h * 8 + j) * ldc + c] = acc[j] * scale;
}

__device__ __forceinline__ v8f zeroC() {
  v8f a = {0.f, 0.f, 0.f, 0.f, 0.f, 0.f, 0.f, 0.f};
  return a;
}

// ---------------------------------------------------------------------------
// Tensor Data Mover: 2D f32 tile (tile_d0 elems/row, tile_d1 rows, row stride
// stride0 elems) from global into LDS. Issue from ONE wave only; follow with
// s_wait_tensorcnt(0) in that wave + __syncthreads().
// ---------------------------------------------------------------------------
#if defined(__has_builtin)
#if __has_builtin(__builtin_amdgcn_tensor_load_to_lds)
#define HAVE_TDM 1
#endif
#endif

#ifdef HAVE_TDM
__device__ __forceinline__ void tdm_load_2d(void* ldsDst, const void* gsrc,
                                            unsigned tile_d0, unsigned tile_d1,
                                            unsigned stride0) {
  unsigned long long ga = (unsigned long long)(size_t)gsrc;
  unsigned ldsoff = (unsigned)(size_t)ldsDst;  // flat addr low 32 bits == LDS offset
  v4u g0;
  g0[0] = 1u;                                      // D#: count=1 (valid), user mode
  g0[1] = ldsoff;                                  // D#.lds_addr
  g0[2] = (unsigned)(ga & 0xffffffffull);          // D#.global_addr[31:0]
  g0[3] = (unsigned)((ga >> 32) & 0x01ffffffull)   // D#.global_addr[56:32]
          | (2u << 30);                            // D#.type = 2 ("image")
  v8i g1;
  g1[0] = (int)(2u << 16);                         // data_size = 4 bytes
  g1[1] = (int)((tile_d0 & 0xffffu) << 16);        // tensor_dim0 lo16
  g1[2] = (int)((tile_d0 >> 16) | (tile_d1 << 16)); // tensor_dim0 hi16 | tensor_dim1 lo16
  g1[3] = (int)((tile_d0 & 0xffffu) << 16);        // tensor_dim1 hi16=0 | tile_dim0
  g1[4] = (int)(tile_d1 & 0xffffu);                // tile_dim1 | tile_dim2 = 0
  g1[5] = (int)stride0;                            // tensor_dim0_stride lo32
  g1[6] = 0;                                       // stride0 hi16 | tensor_dim1_stride lo16
  g1[7] = 0;
  v4i gz = {0, 0, 0, 0};
#if __clang_major__ >= 23
  v8i gz8 = {0, 0, 0, 0, 0, 0, 0, 0};
  __builtin_amdgcn_tensor_load_to_lds(g0, g1, gz, gz, gz8, 0);
#else
  __builtin_amdgcn_tensor_load_to_lds(g0, g1, gz, gz, 0);
#endif
}
#endif

// ---------------------------------------------------------------------------
// Kernel 1: short causal attention. One WG per sequence n (256 WGs, 128 thr).
// ---------------------------------------------------------------------------
__global__ void __launch_bounds__(128) short_attn_kernel(
    const float* __restrict__ Ht, const float* __restrict__ H,
    const float* __restrict__ sq_w, const float* __restrict__ sk_w,
    const float* __restrict__ sv_w, const float* __restrict__ s_in_b,
    const float* __restrict__ s_out_w, const float* __restrict__ s_out_b,
    float* __restrict__ outHt, float* __restrict__ outH) {
  extern __shared__ float sm[];
  float* Htl = sm;           // 64x128 = 8192   (P aliases first 4096 later)
  float* Hl  = sm + 8192;    // 64x256 = 16384  (O aliases first 8192 later)
  float* Ql  = sm + 24576;   // 64x128
  float* Kl  = sm + 32768;   // 64x128
  float* Vl  = sm + 40960;   // 64x128
  float* Pl  = Htl;          // 64x64
  float* Ol  = Hl;           // 64x128

  const int n = blockIdx.x, tid = threadIdx.x;
  const int lane = tid & 31, wave = tid >> 5;

#ifdef HAVE_TDM
  if (tid < 32) {
    tdm_load_2d(Htl, Ht + (size_t)n * 8192, 8192, 1, 8192);
    tdm_load_2d(Hl,  H  + (size_t)n * 16384, 16384, 1, 16384);
    __builtin_amdgcn_s_wait_tensorcnt(0);
  }
#else
  {
    const float4* s4 = (const float4*)(Ht + (size_t)n * 8192);
    float4* d4 = (float4*)Htl;
    for (int i = tid; i < 2048; i += 128) d4[i] = s4[i];
    const float4* h4 = (const float4*)(H + (size_t)n * 16384);
    float4* g4 = (float4*)Hl;
    for (int i = tid; i < 4096; i += 128) g4[i] = h4[i];
  }
#endif
  __syncthreads();

  // Q = Ht @ sq_w^T + bq   (64x128, K=128)
  for (int tile = wave; tile < 32; tile += 4) {
    int rt = tile >> 3, ct = tile & 7;
    v8f acc; float bias = s_in_b[ct * 16 + (lane & 15)];
#pragma unroll
    for (int j = 0; j < 8; ++j) acc[j] = bias;
    for (int kk = 0; kk < 4; ++kk)
      acc = wmma3(loadA(Htl + rt * 16 * 128 + kk * 32, 128, lane),
                  loadBT(sq_w + (ct * 16) * 128 + kk * 32, 128, lane), acc);
    storeC(Ql + rt * 16 * 128 + ct * 16, 128, lane, acc, 1.0f);
  }
  // K = H @ sk_w^T + bk   (64x128, K=256)
  for (int tile = wave; tile < 32; tile += 4) {
    int rt = tile >> 3, ct = tile & 7;
    v8f acc; float bias = s_in_b[128 + ct * 16 + (lane & 15)];
#pragma unroll
    for (int j = 0; j < 8; ++j) acc[j] = bias;
    for (int kk = 0; kk < 8; ++kk)
      acc = wmma3(loadA(Hl + rt * 16 * 256 + kk * 32, 256, lane),
                  loadBT(sk_w + (ct * 16) * 256 + kk * 32, 256, lane), acc);
    storeC(Kl + rt * 16 * 128 + ct * 16, 128, lane, acc, 1.0f);
  }
  // V = H @ sv_w^T + bv
  for (int tile = wave; tile < 32; tile += 4) {
    int rt = tile >> 3, ct = tile & 7;
    v8f acc; float bias = s_in_b[256 + ct * 16 + (lane & 15)];
#pragma unroll
    for (int j = 0; j < 8; ++j) acc[j] = bias;
    for (int kk = 0; kk < 8; ++kk)
      acc = wmma3(loadA(Hl + rt * 16 * 256 + kk * 32, 256, lane),
                  loadBT(sv_w + (ct * 16) * 256 + kk * 32, 256, lane), acc);
    storeC(Vl + rt * 16 * 128 + ct * 16, 128, lane, acc, 1.0f);
  }
  __syncthreads();

  // P = Q @ K^T / sqrt(D)   (64x64, K=128); Kl row-major so B(k,c)=Kl[c*128+k]
  for (int tile = wave; tile < 16; tile += 4) {
    int rt = tile >> 2, ct = tile & 3;
    v8f acc = zeroC();
    for (int kk = 0; kk < 4; ++kk)
      acc = wmma3(loadA(Ql + rt * 16 * 128 + kk * 32, 128, lane),
                  loadBT(Kl + (ct * 16) * 128 + kk * 32, 128, lane), acc);
    storeC(Pl + rt * 16 * 64 + ct * 16, 64, lane, acc, 0.08838834764831845f);
  }
  __syncthreads();

  // causal softmax rows
  if (tid < 64) {
    float* row = Pl + tid * 64;
    float mx = -1e30f;
    for (int j = 0; j <= tid; ++j) mx = fmaxf(mx, row[j]);
    float sum = 0.f;
    for (int j = 0; j <= tid; ++j) { float e = expf(row[j] - mx); row[j] = e; sum += e; }
    float inv = 1.0f / sum;
    for (int j = 0; j < 64; ++j) row[j] = (j <= tid) ? row[j] * inv : 0.0f;
  }
  __syncthreads();

  // O = P @ V   (64x128, K=64); Vl is K-major for this product
  for (int tile = wave; tile < 32; tile += 4) {
    int rt = tile >> 3, ct = tile & 7;
    v8f acc = zeroC();
    for (int kk = 0; kk < 2; ++kk)
      acc = wmma3(loadA(Pl + rt * 16 * 64 + kk * 32, 64, lane),
                  loadB(Vl + (kk * 32) * 128 + ct * 16, 128, lane), acc);
    storeC(Ol + rt * 16 * 128 + ct * 16, 128, lane, acc, 1.0f);
  }
  __syncthreads();

  // new_Ht = O @ s_out_w^T + b; write to outHt and first half of outH
  for (int tile = wave; tile < 32; tile += 4) {
    int rt = tile >> 3, ct = tile & 7;
    v8f acc; float bias = s_out_b[ct * 16 + (lane & 15)];
#pragma unroll
    for (int j = 0; j < 8; ++j) acc[j] = bias;
    for (int kk = 0; kk < 4; ++kk)
      acc = wmma3(loadA(Ol + rt * 16 * 128 + kk * 32, 128, lane),
                  loadBT(s_out_w + (ct * 16) * 128 + kk * 32, 128, lane), acc);
    int c = lane & 15, h = lane >> 4;
#pragma unroll
    for (int j = 0; j < 8; ++j) {
      int t = rt * 16 + h * 8 + j, d = ct * 16 + c;
      float v = acc[j];
      outHt[(size_t)n * 8192 + t * 128 + d] = v;
      outH [(size_t)n * 16384 + t * 256 + d] = v;
    }
  }

  // second half of new_H: shifted Ht (zero when n % B == 0)
  {
    bool zero = (n % BB) == 0;
    const float4* prev = (const float4*)(Ht + (size_t)(n - 1) * 8192);
    for (int i = tid; i < 2048; i += 128) {
      float4 v = {0.f, 0.f, 0.f, 0.f};
      if (!zero) v = prev[i];
      int t = i >> 5, d4 = i & 31;
      ((float4*)(outH + (size_t)n * 16384 + t * 256 + 128))[d4] = v;
    }
  }
}

// ---------------------------------------------------------------------------
// Kernel 2: long attention vs memory slots. One WG per n (256 WGs, 128 thr).
// ---------------------------------------------------------------------------
__global__ void __launch_bounds__(128) long_attn_kernel(
    const float* __restrict__ Hh, const float* __restrict__ Memb,
    const float* __restrict__ l_in_w, const float* __restrict__ l_in_b,
    const float* __restrict__ l_out_w, const float* __restrict__ l_out_b,
    float* __restrict__ outHh) {
  extern __shared__ float sm[];
  float* Hhl = sm;           // 64x128 = 8192 (O aliases later)
  float* Ml  = sm + 8192;    // 32x128 = 4096 (P 64x32=2048 aliases later)
  float* Ql  = sm + 12288;   // 8192
  float* Kl  = sm + 20480;   // 4096
  float* Vl  = sm + 24576;   // 4096
  float* Pl  = Ml;
  float* Ol  = Hhl;

  const int n = blockIdx.x, tid = threadIdx.x;
  const int lane = tid & 31, wave = tid >> 5;
  const int bidx = n >> 3;  // n = b*S + s

#ifdef HAVE_TDM
  if (tid < 32) {
    tdm_load_2d(Hhl, Hh + (size_t)n * 8192, 8192, 1, 8192);
    tdm_load_2d(Ml,  Memb + (size_t)bidx * 4096, 4096, 1, 4096);
    __builtin_amdgcn_s_wait_tensorcnt(0);
  }
#else
  {
    const float4* s4 = (const float4*)(Hh + (size_t)n * 8192);
    float4* d4 = (float4*)Hhl;
    for (int i = tid; i < 2048; i += 128) d4[i] = s4[i];
    const float4* m4 = (const float4*)(Memb + (size_t)bidx * 4096);
    float4* e4 = (float4*)Ml;
    for (int i = tid; i < 1024; i += 128) e4[i] = m4[i];
  }
#endif
  __syncthreads();

  // Q = Hh @ lq^T + lbq (lq = l_in_w rows 0..127)
  for (int tile = wave; tile < 32; tile += 4) {
    int rt = tile >> 3, ct = tile & 7;
    v8f acc; float bias = l_in_b[ct * 16 + (lane & 15)];
#pragma unroll
    for (int j = 0; j < 8; ++j) acc[j] = bias;
    for (int kk = 0; kk < 4; ++kk)
      acc = wmma3(loadA(Hhl + rt * 16 * 128 + kk * 32, 128, lane),
                  loadBT(l_in_w + (ct * 16) * 128 + kk * 32, 128, lane), acc);
    storeC(Ql + rt * 16 * 128 + ct * 16, 128, lane, acc, 1.0f);
  }
  // K = M @ lk^T + lbk  (32x128); V = M @ lv^T + lbv
  for (int tile = wave; tile < 16; tile += 4) {
    int rt = tile >> 3, ct = tile & 7;
    v8f acc; float bias = l_in_b[128 + ct * 16 + (lane & 15)];
#pragma unroll
    for (int j = 0; j < 8; ++j) acc[j] = bias;
    for (int kk = 0; kk < 4; ++kk)
      acc = wmma3(loadA(Ml + rt * 16 * 128 + kk * 32, 128, lane),
                  loadBT(l_in_w + (128 + ct * 16) * 128 + kk * 32, 128, lane), acc);
    storeC(Kl + rt * 16 * 128 + ct * 16, 128, lane, acc, 1.0f);
  }
  for (int tile = wave; tile < 16; tile += 4) {
    int rt = tile >> 3, ct = tile & 7;
    v8f acc; float bias = l_in_b[256 + ct * 16 + (lane & 15)];
#pragma unroll
    for (int j = 0; j < 8; ++j) acc[j] = bias;
    for (int kk = 0; kk < 4; ++kk)
      acc = wmma3(loadA(Ml + rt * 16 * 128 + kk * 32, 128, lane),
                  loadBT(l_in_w + (256 + ct * 16) * 128 + kk * 32, 128, lane), acc);
    storeC(Vl + rt * 16 * 128 + ct * 16, 128, lane, acc, 1.0f);
  }
  __syncthreads();

  // P = Q @ K^T / sqrt(D)  (64x32)
  for (int tile = wave; tile < 8; tile += 4) {
    int rt = tile >> 1, ct = tile & 1;
    v8f acc = zeroC();
    for (int kk = 0; kk < 4; ++kk)
      acc = wmma3(loadA(Ql + rt * 16 * 128 + kk * 32, 128, lane),
                  loadBT(Kl + (ct * 16) * 128 + kk * 32, 128, lane), acc);
    storeC(Pl + rt * 16 * 32 + ct * 16, 32, lane, acc, 0.08838834764831845f);
  }
  __syncthreads();

  if (tid < 64) {
    float* row = Pl + tid * 32;
    float mx = -1e30f;
    for (int j = 0; j < 32; ++j) mx = fmaxf(mx, row[j]);
    float sum = 0.f;
    for (int j = 0; j < 32; ++j) { float e = expf(row[j] - mx); row[j] = e; sum += e; }
    float inv = 1.0f / sum;
    for (int j = 0; j < 32; ++j) row[j] *= inv;
  }
  __syncthreads();

  // O = P @ V  (64x128, K=32)
  for (int tile = wave; tile < 32; tile += 4) {
    int rt = tile >> 3, ct = tile & 7;
    v8f acc = zeroC();
    acc = wmma3(loadA(Pl + rt * 16 * 32, 32, lane),
                loadB(Vl + ct * 16, 128, lane), acc);
    storeC(Ol + rt * 16 * 128 + ct * 16, 128, lane, acc, 1.0f);
  }
  __syncthreads();

  // new_Hh = O @ l_out_w^T + l_out_b
  for (int tile = wave; tile < 32; tile += 4) {
    int rt = tile >> 3, ct = tile & 7;
    v8f acc; float bias = l_out_b[ct * 16 + (lane & 15)];
#pragma unroll
    for (int j = 0; j < 8; ++j) acc[j] = bias;
    for (int kk = 0; kk < 4; ++kk)
      acc = wmma3(loadA(Ol + rt * 16 * 128 + kk * 32, 128, lane),
                  loadBT(l_out_w + (ct * 16) * 128 + kk * 32, 128, lane), acc);
    int c = lane & 15, h = lane >> 4;
#pragma unroll
    for (int j = 0; j < 8; ++j) {
      int t = rt * 16 + h * 8 + j, d = ct * 16 + c;
      outHh[(size_t)n * 8192 + t * 128 + d] = acc[j];
    }
  }
}

// ---------------------------------------------------------------------------
// Kernel 3 (per step): Wc[:,i,t,:] = cat[:,t,:] @ W[i,t]^T  — 32x128x128 GEMM
// per (i,t). grid = (96 t, 32 i), 128 threads. W streamed exactly once/step.
// A-tile fetched as a 2D strided TDM tile (128 x 32 rows).
// ---------------------------------------------------------------------------
__global__ void __launch_bounds__(128) wc_gemm_kernel(
    const float* __restrict__ W, const float* __restrict__ curM,
    const float* __restrict__ Ht, float* __restrict__ Wc, int step) {
  extern __shared__ float sm[];  // 32x128 A tile
  const int t = blockIdx.x, i = blockIdx.y;
  const int tid = threadIdx.x, lane = tid & 31, wave = tid >> 5;

#ifdef HAVE_TDM
  if (tid < 32) {
    if (t < MMEM)
      tdm_load_2d(sm, curM + (size_t)t * 128, 128, 32, 4096);       // rows b: stride M*D
    else
      tdm_load_2d(sm, Ht + (size_t)(step * TT + (t - MMEM)) * 128,  // rows b: stride S*T*D
                  128, 32, 65536);
    __builtin_amdgcn_s_wait_tensorcnt(0);
  }
#else
  {
    float4* a4 = (float4*)sm;
    if (t < MMEM) {
      for (int idx = tid; idx < 1024; idx += 128) {
        int b = idx >> 5, k4 = idx & 31;
        a4[idx] = ((const float4*)(curM + (size_t)(b * MMEM + t) * 128))[k4];
      }
    } else {
      for (int idx = tid; idx < 1024; idx += 128) {
        int b = idx >> 5, k4 = idx & 31;
        a4[idx] = ((const float4*)(Ht + (size_t)((b * SS + step) * TT + (t - MMEM)) * 128))[k4];
      }
    }
  }
#endif
  __syncthreads();

  const float* Wblk = W + (size_t)(i * MT + t) * 16384;  // W[i,t] 128x128 row-major [d][k]
#pragma unroll
  for (int rt = 0; rt < 2; ++rt) {
    Split16 a0 = loadA(sm + rt * 2048 +  0, 128, lane);
    Split16 a1 = loadA(sm + rt * 2048 + 32, 128, lane);
    Split16 a2 = loadA(sm + rt * 2048 + 64, 128, lane);
    Split16 a3 = loadA(sm + rt * 2048 + 96, 128, lane);
#pragma unroll
    for (int cc = 0; cc < 2; ++cc) {
      int ct = wave * 2 + cc;
      const float* Bb = Wblk + (ct * 16) * 128;
      __builtin_prefetch(Bb + 2048, 0, 1);
      v8f acc = zeroC();
      acc = wmma3(a0, loadBT(Bb +  0, 128, lane), acc);
      acc = wmma3(a1, loadBT(Bb + 32, 128, lane), acc);
      acc = wmma3(a2, loadBT(Bb + 64, 128, lane), acc);
      acc = wmma3(a3, loadBT(Bb + 96, 128, lane), acc);
      int c = lane & 15, h = lane >> 4;
      size_t col = (size_t)(i * MT + t) * 128 + ct * 16 + c;
#pragma unroll
      for (int j = 0; j < 8; ++j) {
        int b = rt * 16 + h * 8 + j;
        Wc[(size_t)b * (MMEM * MT * DD) + col] = acc[j];  // layout [b][i][t][d]
      }
    }
  }
}

// ---------------------------------------------------------------------------
// Kernel 4 (per step): power iterations on LDS-resident Wc[b,i] (48 KB, TDM),
// JAX threefry2x32 PRNG init. grid = 1024 (b*32+i), 128 threads.
// ---------------------------------------------------------------------------
__device__ __forceinline__ unsigned rotl32(unsigned x, int n) {
  return (x << n) | (x >> (32 - n));
}
__device__ __forceinline__ void threefry2x32(unsigned k0, unsigned k1,
                                             unsigned x0, unsigned x1,
                                             unsigned& o0, unsigned& o1) {
  unsigned ks2 = 0x1BD11BDAu ^ k0 ^ k1;
  x0 += k0; x1 += k1;
#define TFR(r) { x0 += x1; x1 = rotl32(x1, r); x1 ^= x0; }
  TFR(13) TFR(15) TFR(26) TFR(6)  x0 += k1;  x1 += ks2 + 1u;
  TFR(17) TFR(29) TFR(16) TFR(24) x0 += ks2; x1 += k0 + 2u;
  TFR(13) TFR(15) TFR(26) TFR(6)  x0 += k0;  x1 += k1 + 3u;
  TFR(17) TFR(29) TFR(16) TFR(24) x0 += k1;  x1 += ks2 + 4u;
  TFR(13) TFR(15) TFR(26) TFR(6)  x0 += ks2; x1 += k0 + 5u;
#undef TFR
  o0 = x0; o1 = x1;
}

__global__ void __launch_bounds__(128) dm_iter_kernel(
    const float* __restrict__ Wc, float* __restrict__ dstM,
    float* __restrict__ outM, int step) {
  __shared__ float smW[MT * DD];  // 96x128 = 48 KB
  __shared__ float newv[DD];
  __shared__ float bvec[MT];
  __shared__ float red;

  const int bi = blockIdx.x, tid = threadIdx.x;
  const int b = bi >> 5, i = bi & 31;
  const float* srcf = Wc + (size_t)b * (MMEM * MT * DD) + (size_t)i * (MT * DD);

#ifdef HAVE_TDM
  if (tid < 32) {
    tdm_load_2d(smW, srcf, 12288, 1, 12288);
    __builtin_amdgcn_s_wait_tensorcnt(0);
  }
#else
  {
    const float4* src = (const float4*)srcf;
    float4* d4 = (float4*)smW;
    for (int idx = tid; idx < 3072; idx += 128) d4[idx] = src[idx];
  }
#endif

  // init new = jax.random.uniform(fold_in(key(1), step), (B,M,D))
  {
    unsigned fk0, fk1;
    threefry2x32(0u, 1u, 0u, (unsigned)step, fk0, fk1);  // fold_in
    unsigned j = (unsigned)(bi * 128 + tid);
    unsigned o0, o1, bits;
    if (j < 65536u) { threefry2x32(fk0, fk1, j, j + 65536u, o0, o1); bits = o0; }
    else            { threefry2x32(fk0, fk1, j - 65536u, j, o0, o1); bits = o1; }
    newv[tid] = __uint_as_float((bits >> 9) | 0x3F800000u) - 1.0f;
  }
  __syncthreads();

  for (int it = 0; it < 3; ++it) {
    // b_t = dot(new, Wc[t,:]); e = exp(b_t)  (reference applies no max-shift)
    if (tid < MT) {
      float acc = 0.f;
      const float* row = smW + tid * DD;
      for (int dd = 0; dd < DD; ++dd) { int d = (dd + tid) & 127; acc += newv[d] * row[d]; }
      bvec[tid] = expf(acc);
    }
    __syncthreads();
    if (tid == 0) {
      float s = 0.f;
      for (int t2 = 0; t2 < MT; ++t2) s += bvec[t2];
      red = 1.0f / s;
    }
    __syncthreads();
    float inv = red;
    float sd = 0.f;
    for (int t2 = 0; t2 < MT; ++t2) sd += bvec[t2] * smW[t2 * DD + tid];
    sd = tanhf(sd * inv);
    __syncthreads();
    newv[tid] = sd * sd;
    __syncthreads();
    if (tid == 0) {
      float s = 0.f;
      for (int d = 0; d < DD; ++d) s += newv[d];
      red = sqrtf(s) + 1e-4f;
    }
    __syncthreads();
    float nn = red;
    float nv = sd * nn / (1.0f + nn * nn);  // (n^2/(1+n^2)) * (s/n)
    __syncthreads();
    newv[tid] = nv;
    __syncthreads();
  }

  float val = newv[tid];
  if (step < SS - 1) dstM[(size_t)bi * 128 + tid] = val;
  else               outM[(size_t)bi * 128 + tid] = val;
}

// ---------------------------------------------------------------------------
extern "C" void kernel_launch(void* const* d_in, const int* in_sizes, int n_in,
                              void* d_out, int out_size, void* d_ws, size_t ws_size,
                              hipStream_t stream) {
  (void)in_sizes; (void)n_in; (void)out_size; (void)ws_size;
  const float* Ht     = (const float*)d_in[0];
  const float* H      = (const float*)d_in[1];
  const float* Hh     = (const float*)d_in[2];
  const float* Memb   = (const float*)d_in[3];
  const float* W      = (const float*)d_in[4];
  const float* sq_w   = (const float*)d_in[5];
  const float* sk_w   = (const float*)d_in[6];
  const float* sv_w   = (const float*)d_in[7];
  const float* s_in_b = (const float*)d_in[8];
  const float* s_out_w= (const float*)d_in[9];
  const float* s_out_b= (const float*)d_in[10];
  const float* l_in_w = (const float*)d_in[11];
  const float* l_in_b = (const float*)d_in[12];
  const float* l_out_w= (const float*)d_in[13];
  const float* l_out_b= (const float*)d_in[14];

  float* out   = (float*)d_out;
  float* outHt = out;                         // B*S*T*D    = 2097152
  float* outH  = outHt + 2097152;             // B*S*T*2D   = 4194304
  float* outHh = outH  + 4194304;             // B*S*T*D    = 2097152
  float* outM  = outHh + 2097152;             // B*M*D      = 131072

  float* wsWc = (float*)d_ws;                 // 32*32*96*128 floats (48 MB)
  float* wsM0 = wsWc + (size_t)BB * MMEM * MT * DD;
  float* wsM1 = wsM0 + (size_t)BB * MMEM * DD;
  float* bufs[2] = { wsM0, wsM1 };

  short_attn_kernel<<<NSEQ, 128, 49152 * sizeof(float), stream>>>(
      Ht, H, sq_w, sk_w, sv_w, s_in_b, s_out_w, s_out_b, outHt, outH);
  long_attn_kernel<<<NSEQ, 128, 28672 * sizeof(float), stream>>>(
      Hh, Memb, l_in_w, l_in_b, l_out_w, l_out_b, outHh);

  for (int step = 0; step < SS; ++step) {
    const float* cur = (step == 0) ? Memb : bufs[step & 1];
    wc_gemm_kernel<<<dim3(MT, MMEM), 128, 4096 * sizeof(float), stream>>>(
        W, cur, Ht, wsWc, step);
    dm_iter_kernel<<<BB * MMEM, 128, 0, stream>>>(
        wsWc, bufs[(step + 1) & 1], outM, step);
  }
}
